// NNApprox_68796786147684
// MI455X (gfx1250) — compile-verified, matrix operand order checked by reading, not run on hardware
//
#include <hip/hip_runtime.h>
#include <hip/hip_bf16.h>
#include <math.h>

// ---------------------------------------------------------------------------
// CDNA5 (gfx1250) bf16 WMMA types
// ---------------------------------------------------------------------------
typedef __attribute__((ext_vector_type(16))) __bf16 v16bf;
typedef __attribute__((ext_vector_type(8)))  float  v8f;

union FragU { uint4 u[2]; v16bf v; };

__device__ __forceinline__ unsigned f2bf1(float f) {
  union { float f; unsigned u; } c; c.f = f;
  unsigned u = c.u;
  unsigned r = u + 0x7FFFu + ((u >> 16) & 1u);   // round-to-nearest-even
  return r >> 16;
}

#if __has_builtin(__builtin_amdgcn_cvt_pk_bf16_f32)
__device__ __forceinline__ unsigned pack2(float lo, float hi) {
  auto v = __builtin_amdgcn_cvt_pk_bf16_f32(lo, hi);  // v_cvt_pk_bf16_f32
  unsigned r; __builtin_memcpy(&r, &v, 4);
  return r;
}
#else
__device__ __forceinline__ unsigned pack2(float lo, float hi) {
  return f2bf1(lo) | (f2bf1(hi) << 16);
}
#endif

__device__ __forceinline__ float siluf(float x) {
  return x * (1.0f / (1.0f + __expf(-x)));
}

__device__ __forceinline__ void wait_async0() {
#if __has_builtin(__builtin_amdgcn_s_wait_asynccnt)
  __builtin_amdgcn_s_wait_asynccnt(0);
#else
  asm volatile("s_wait_asynccnt 0" ::: "memory");
#endif
}

// Async copy 16B global -> LDS (GLOBAL_LOAD_ASYNC_TO_LDS_B128, ASYNCcnt).
__device__ __forceinline__ void async_copy_b128(unsigned lds_addr,
                                                const void* gptr) {
  asm volatile("global_load_async_to_lds_b128 %0, %1, off"
               :
               : "v"(lds_addr), "v"((unsigned long long)(uintptr_t)gptr)
               : "memory");
}

// Load one 16x32 bf16 operand fragment from an LDS tile stored [row][k](BK=64).
// Per ISA 16-bit A/B layout: lane-group g covers K chunks [8g..8g+7] and
// [16+8g..16+8g+7] of the 32-wide K slice -> two contiguous b128 LDS reads.
__device__ __forceinline__ v16bf lds_frag(const unsigned short* tile, int row,
                                          int k0, int g) {
  const unsigned short* p = tile + row * 64 + k0 + 8 * g;
  FragU f;
  f.u[0] = *(const uint4*)(p);
  f.u[1] = *(const uint4*)(p + 16);
  return f.v;
}

__device__ __forceinline__ v8f wmma_bf16(v16bf a, v16bf b, v8f c) {
  return __builtin_amdgcn_wmma_f32_16x16x32_bf16(false, a, false, b,
                                                 (short)0, c, false, false);
}

// ---------------------------------------------------------------------------
// Generic GEMM: C(MxN) = A(bf16, MxK, row-major) * B(f32 weights, NxK)^T
// BM=128 BN=128 BK=64, 256 threads = 8 waves (2x4), wave tile 64x32.
// A tile: async global->LDS (bf16 passthrough). B tile: reg-staged + cvt_pk.
// EPI: 0 = f32 store, 1 = bf16 store, 2 = bias+silu+bf16, 3 = bias+f32
// ---------------------------------------------------------------------------
template <int EPI>
__global__ __launch_bounds__(256)
void gemm_bf16A_f32B(const unsigned short* __restrict__ A, int lda,
                     const float* __restrict__ Bw, int ldb,
                     void* __restrict__ Cp, int ldc,
                     const float* __restrict__ bias, int Kdim) {
  constexpr int BM = 128, BN = 128, BK = 64;
  __shared__ unsigned short As[2][BM * BK];
  __shared__ unsigned short Bs[2][BN * BK];

  const int tid    = threadIdx.x;
  const int lane   = tid & 31;
  const int waveId = tid >> 5;
  const int wm     = waveId & 1;   // 0..1 : wave M position (64 rows each)
  const int wn     = waveId >> 1;  // 0..3 : wave N position (32 cols each)
  const int g      = lane >> 4;
  const int ln     = lane & 15;
  const long blockM = (long)blockIdx.y * BM;
  const long blockN = (long)blockIdx.x * BN;

  const int aRow0 = tid >> 3, aCol = (tid & 7) * 8;    // bf16 A: b128 chunks
  const int bRow0 = tid >> 4, bCol = (tid & 15) * 4;   // f32 B: float4 chunks

  const int numK = Kdim / BK;

  float4 bReg[8];

  auto asyncLoadA = [&](int t, int buf) {
#pragma unroll
    for (int r = 0; r < 4; ++r) {
      const unsigned lds =
          (unsigned)(uintptr_t)&As[buf][(aRow0 + 32 * r) * BK + aCol];
      const unsigned short* p =
          A + (blockM + aRow0 + 32 * r) * lda + (long)t * BK + aCol;
      async_copy_b128(lds, p);
    }
  };
  auto loadB = [&](int t) {
#pragma unroll
    for (int r = 0; r < 8; ++r) {
      const float* p = Bw + (blockN + bRow0 + 16 * r) * ldb + (long)t * BK + bCol;
      bReg[r] = *(const float4*)p;
    }
  };
  auto storeB = [&](int buf) {
#pragma unroll
    for (int r = 0; r < 8; ++r) {
      uint2 v;
      v.x = pack2(bReg[r].x, bReg[r].y);
      v.y = pack2(bReg[r].z, bReg[r].w);
      *(uint2*)&Bs[buf][(bRow0 + 16 * r) * BK + bCol] = v;
    }
  };

  v8f acc[4][2] = {};

  asyncLoadA(0, 0);
  loadB(0);
  storeB(0);
  wait_async0();
  __syncthreads();

  for (int t = 0; t < numK; ++t) {
    const int cur = t & 1;
    if (t + 1 < numK) {
      loadB(t + 1);
      asyncLoadA(t + 1, cur ^ 1);
    }
    if (t + 2 < numK) {
      __builtin_prefetch(A + (blockM + aRow0) * lda + (long)(t + 2) * BK + aCol, 0, 1);
      __builtin_prefetch(Bw + (blockN + bRow0) * ldb + (long)(t + 2) * BK + bCol, 0, 1);
    }
    const unsigned short* At = As[cur];
    const unsigned short* Bt = Bs[cur];
#pragma unroll
    for (int kk = 0; kk < 2; ++kk) {
      const int k0 = kk * 32;
      v16bf a[4], b[2];
#pragma unroll
      for (int i = 0; i < 4; ++i) a[i] = lds_frag(At, wm * 64 + i * 16 + ln, k0, g);
#pragma unroll
      for (int j = 0; j < 2; ++j) b[j] = lds_frag(Bt, wn * 32 + j * 16 + ln, k0, g);
#pragma unroll
      for (int i = 0; i < 4; ++i)
#pragma unroll
        for (int j = 0; j < 2; ++j) acc[i][j] = wmma_bf16(a[i], b[j], acc[i][j]);
    }
    if (t + 1 < numK) storeB(cur ^ 1);
    wait_async0();
    __syncthreads();
  }

  // Epilogue. C/D layout: VGPR v, lane L -> M = v + 8*(L/16), N = L%16.
  const long baseM = blockM + wm * 64;
  const long baseN = blockN + wn * 32;
#pragma unroll
  for (int j = 0; j < 2; ++j) {
    const long col = baseN + j * 16 + ln;
    float bv = 0.0f;
    if (EPI == 2 || EPI == 3) bv = bias[col];
#pragma unroll
    for (int i = 0; i < 4; ++i) {
#pragma unroll
      for (int v = 0; v < 8; ++v) {
        const long row = baseM + i * 16 + 8 * g + v;
        float val = acc[i][j][v];
        if (EPI == 0) {
          ((float*)Cp)[row * ldc + col] = val;
        } else if (EPI == 1) {
          ((unsigned short*)Cp)[row * ldc + col] = (unsigned short)f2bf1(val);
        } else if (EPI == 2) {
          float s = siluf(val + bv);
          ((unsigned short*)Cp)[row * ldc + col] = (unsigned short)f2bf1(s);
        } else {
          ((float*)Cp)[row * ldc + col] = val + bv;
        }
      }
    }
  }
}

// ---------------------------------------------------------------------------
// Stage 1: G = silu(X*Wg^T) * (X*Wu^T), bf16 out. X f32 (MxK), Wg/Wu f32 (NxK).
// BM=128 BN=64 BK=64; one shared X tile drives both weight streams.
// ---------------------------------------------------------------------------
__global__ __launch_bounds__(256)
void gemm_dual_silu(const float* __restrict__ X, int lda,
                    const float* __restrict__ Wg, const float* __restrict__ Wu,
                    int ldb, unsigned short* __restrict__ G, int ldc, int Kdim) {
  constexpr int BM = 128, BN = 64, BK = 64;
  __shared__ unsigned short As[2][BM * BK];
  __shared__ unsigned short Bgs[2][BN * BK];
  __shared__ unsigned short Bus[2][BN * BK];

  const int tid    = threadIdx.x;
  const int lane   = tid & 31;
  const int waveId = tid >> 5;
  const int wm     = waveId & 1;   // 64-row halves
  const int wn     = waveId >> 1;  // 0..3, 16 cols each
  const int g      = lane >> 4;
  const int ln     = lane & 15;
  const long blockM = (long)blockIdx.y * BM;
  const long blockN = (long)blockIdx.x * BN;

  const int row0 = tid >> 4, col0 = (tid & 15) * 4;  // float4 staging
  const int numK = Kdim / BK;

  float4 aReg[8], bgReg[4], buReg[4];

  auto loadAll = [&](int t) {
#pragma unroll
    for (int r = 0; r < 8; ++r)
      aReg[r] = *(const float4*)(X + (blockM + row0 + 16 * r) * lda + (long)t * BK + col0);
#pragma unroll
    for (int r = 0; r < 4; ++r) {
      bgReg[r] = *(const float4*)(Wg + (blockN + row0 + 16 * r) * ldb + (long)t * BK + col0);
      buReg[r] = *(const float4*)(Wu + (blockN + row0 + 16 * r) * ldb + (long)t * BK + col0);
    }
  };
  auto storeAll = [&](int buf) {
#pragma unroll
    for (int r = 0; r < 8; ++r) {
      uint2 v; v.x = pack2(aReg[r].x, aReg[r].y); v.y = pack2(aReg[r].z, aReg[r].w);
      *(uint2*)&As[buf][(row0 + 16 * r) * BK + col0] = v;
    }
#pragma unroll
    for (int r = 0; r < 4; ++r) {
      uint2 v; v.x = pack2(bgReg[r].x, bgReg[r].y); v.y = pack2(bgReg[r].z, bgReg[r].w);
      *(uint2*)&Bgs[buf][(row0 + 16 * r) * BK + col0] = v;
      uint2 w; w.x = pack2(buReg[r].x, buReg[r].y); w.y = pack2(buReg[r].z, buReg[r].w);
      *(uint2*)&Bus[buf][(row0 + 16 * r) * BK + col0] = w;
    }
  };

  v8f accg[4] = {};
  v8f accu[4] = {};

  loadAll(0); storeAll(0);
  __syncthreads();

  for (int t = 0; t < numK; ++t) {
    const int cur = t & 1;
    if (t + 1 < numK) loadAll(t + 1);
    if (t + 2 < numK)
      __builtin_prefetch(X + (blockM + row0) * lda + (long)(t + 2) * BK + col0, 0, 1);
    const unsigned short* At  = As[cur];
    const unsigned short* Bgt = Bgs[cur];
    const unsigned short* But = Bus[cur];
#pragma unroll
    for (int kk = 0; kk < 2; ++kk) {
      const int k0 = kk * 32;
      v16bf a[4];
#pragma unroll
      for (int i = 0; i < 4; ++i) a[i] = lds_frag(At, wm * 64 + i * 16 + ln, k0, g);
      v16bf bg = lds_frag(Bgt, wn * 16 + ln, k0, g);
      v16bf bu = lds_frag(But, wn * 16 + ln, k0, g);
#pragma unroll
      for (int i = 0; i < 4; ++i) {
        accg[i] = wmma_bf16(a[i], bg, accg[i]);
        accu[i] = wmma_bf16(a[i], bu, accu[i]);
      }
    }
    if (t + 1 < numK) storeAll(cur ^ 1);
    __syncthreads();
  }

  const long baseM = blockM + wm * 64;
  const long col   = blockN + wn * 16 + ln;
#pragma unroll
  for (int i = 0; i < 4; ++i) {
#pragma unroll
    for (int v = 0; v < 8; ++v) {
      const long row = baseM + i * 16 + 8 * g + v;
      float gv = siluf(accg[i][v]) * accu[i][v];
      G[row * ldc + col] = (unsigned short)f2bf1(gv);
    }
  }
}

// ---------------------------------------------------------------------------
// Stage 5: per-row L2 norm and mean-abs of (o0 - proj).
// ---------------------------------------------------------------------------
__global__ __launch_bounds__(256)
void rownorm_kernel(const float* __restrict__ o0, const float* __restrict__ pj,
                    float* __restrict__ out, int n, int nrows) {
  const int b = blockIdx.x;
  const float* pa = o0 + (long)b * n;
  const float* pb = pj + (long)b * n;
  float ss = 0.0f, sa = 0.0f;
  for (int i = threadIdx.x; i < n; i += 256) {
    float d = pa[i] - pb[i];
    ss += d * d;
    sa += fabsf(d);
  }
  // wave32 shuffle reduce
#pragma unroll
  for (int off = 16; off > 0; off >>= 1) {
    ss += __shfl_down(ss, off, 32);
    sa += __shfl_down(sa, off, 32);
  }
  __shared__ float sh[2][8];
  const int w = threadIdx.x >> 5;
  if ((threadIdx.x & 31) == 0) { sh[0][w] = ss; sh[1][w] = sa; }
  __syncthreads();
  if (threadIdx.x == 0) {
    float S = 0.0f, A = 0.0f;
#pragma unroll
    for (int i = 0; i < 8; ++i) { S += sh[0][i]; A += sh[1][i]; }
    out[b]         = sqrtf(S);
    out[nrows + b] = A / (float)n;
  }
}

// ---------------------------------------------------------------------------
// Launch
// ---------------------------------------------------------------------------
extern "C" void kernel_launch(void* const* d_in, const int* in_sizes, int n_in,
                              void* d_out, int out_size, void* d_ws, size_t ws_size,
                              hipStream_t stream) {
  (void)in_sizes; (void)n_in; (void)out_size; (void)ws_size;
  const float* x    = (const float*)d_in[0];  // (4096, 4096)
  const float* W1   = (const float*)d_in[1];  // (5, 2048, 4096) -> flat (10240, 4096)
  const float* W3   = (const float*)d_in[2];
  const float* W2   = (const float*)d_in[3];  // (5, 4096, 2048)
  const float* fc1w = (const float*)d_in[4];  // (4096, 16384)
  const float* fc1b = (const float*)d_in[5];
  const float* fc2w = (const float*)d_in[6];  // (4096, 4096)
  const float* fc2b = (const float*)d_in[7];
  float* out = (float*)d_out;                 // [norm(4096) | meanabs(4096)]

  char* ws = (char*)d_ws;
  const size_t GBYTES  = (size_t)4096 * 10240 * 2;  // G bf16 (B, 5*2048)
  const size_t CCBYTES = (size_t)4096 * 16384 * 2;  // concat bf16 (B, 4*4096)
  const size_t O0BYTES = (size_t)4096 * 4096 * 4;   // o0 f32
  unsigned short* G    = (unsigned short*)ws;
  unsigned short* CC   = (unsigned short*)(ws + GBYTES);
  float*          O0   = (float*)(ws + GBYTES + CCBYTES);
  float*          PROJ = (float*)(ws + GBYTES + CCBYTES + O0BYTES);
  unsigned short* H    = (unsigned short*)ws;  // reuse G region after stage 2

  dim3 blk(256);

  // Stage 1: G = silu(x*W1^T) * (x*W3^T)   [M=4096, N=10240, K=4096]
  gemm_dual_silu<<<dim3(10240 / 64, 4096 / 128), blk, 0, stream>>>(
      x, 4096, W1, W3, 4096, G, 10240, 4096);

  // Stage 2: wk[k] = G[:,k] * W2[k]^T      [M=4096, N=4096, K=2048]
  for (int k = 0; k < 5; ++k) {
    const unsigned short* Ak  = G + (size_t)k * 2048;
    const float*          W2k = W2 + (size_t)k * 4096 * 2048;
    if (k == 0) {
      gemm_bf16A_f32B<0><<<dim3(32, 32), blk, 0, stream>>>(
          Ak, 10240, W2k, 2048, O0, 4096, nullptr, 2048);
    } else {
      gemm_bf16A_f32B<1><<<dim3(32, 32), blk, 0, stream>>>(
          Ak, 10240, W2k, 2048, CC + (size_t)(k - 1) * 4096, 16384, nullptr, 2048);
    }
  }

  // Stage 3: H = silu(CC * fc1^T + b)      [M=4096, N=4096, K=16384]
  gemm_bf16A_f32B<2><<<dim3(32, 32), blk, 0, stream>>>(
      CC, 16384, fc1w, 16384, H, 4096, fc1b, 16384);

  // Stage 4: PROJ = H * fc2^T + b          [M=4096, N=4096, K=4096]
  gemm_bf16A_f32B<3><<<dim3(32, 32), blk, 0, stream>>>(
      H, 4096, fc2w, 4096, PROJ, 4096, fc2b, 4096);

  // Stage 5: per-row reductions
  rownorm_kernel<<<4096, blk, 0, stream>>>(O0, PROJ, out, 4096, 4096);
}
